// MultiheadAttentionWithBias_4818953306698
// MI455X (gfx1250) — matrix operand order repaired and checked by reading, not compiled, not run on hardware
//
#include <hip/hip_runtime.h>

#define DIMC  1024
#define NHEAD 16
#define HDIM  64
#define BATCH 4
#define SEQ   2048

typedef __attribute__((ext_vector_type(16))) __bf16 v16bf;
typedef __attribute__((ext_vector_type(8)))  __bf16 v8bf;
typedef __attribute__((ext_vector_type(8)))  float  v8f;

__device__ __forceinline__ __bf16 f2bf(float f) {
  unsigned u = __builtin_bit_cast(unsigned, f);
  unsigned r = 0x7FFFu + ((u >> 16) & 1u);
  unsigned short h = (unsigned short)((u + r) >> 16);
  return __builtin_bit_cast(__bf16, h);
}

__global__ void __launch_bounds__(256) cvt_f32_bf16(const float* __restrict__ s,
                                                    __bf16* __restrict__ d, int n) {
  int i = blockIdx.x * 256 + threadIdx.x;
  if (i < n) d[i] = f2bf(s[i]);
}

// ---------------------------------------------------------------------------
// bf16 GEMM mainloop, wave tile 32x64: acc[2][4] 16x16 tiles.
// A[M,K] row-major, W[N,K] row-major (Linear weight) -> B columns contiguous.
// A fragments: two 16B chunks/lane (contiguous k = 16*h + s);
// B fragments: one 32B chunk/lane  (same contiguous k = 16*h + s).
// ---------------------------------------------------------------------------
__device__ __forceinline__ void gemm_mainloop(const __bf16* __restrict__ A,
                                              const __bf16* __restrict__ W,
                                              int mbase, int nbase, int K,
                                              v8f acc[2][4]) {
  const int lane = threadIdx.x & 31;
  const int l15  = lane & 15;
  const int kh   = lane >> 4;
  const __bf16* ar[2];
  const __bf16* wr[4];
  ar[0] = A + (size_t)(mbase + l15) * K + kh * 16;
  ar[1] = A + (size_t)(mbase + 16 + l15) * K + kh * 16;
  #pragma unroll
  for (int ni = 0; ni < 4; ++ni)
    wr[ni] = W + (size_t)(nbase + ni * 16 + l15) * K + kh * 16;
  #pragma unroll 2
  for (int kk = 0; kk < K; kk += 32) {
    __builtin_prefetch(ar[0] + kk + 256, 0, 3);   // global_prefetch_b8
    __builtin_prefetch(wr[0] + kk + 256, 0, 3);
    v16bf a[2], b[4];
    #pragma unroll
    for (int mi = 0; mi < 2; ++mi) {
      v8bf lo = *(const v8bf*)(ar[mi] + kk);
      v8bf hi = *(const v8bf*)(ar[mi] + kk + 8);
      #pragma unroll
      for (int i = 0; i < 8; ++i) { a[mi][i] = lo[i]; a[mi][8 + i] = hi[i]; }
    }
    #pragma unroll
    for (int ni = 0; ni < 4; ++ni)
      b[ni] = *(const v16bf*)(wr[ni] + kk);
    #pragma unroll
    for (int mi = 0; mi < 2; ++mi)
      #pragma unroll
      for (int ni = 0; ni < 4; ++ni)
        acc[mi][ni] = __builtin_amdgcn_wmma_f32_16x16x32_bf16(
            false, a[mi], false, b[ni], (short)0, acc[mi][ni], false, false);
  }
}

// ---------------------------------------------------------------------------
// QKV projection: [8192,1024] x [3072,1024]^T + bias.
// q,k scattered to [B,H,T,D] bf16; v scattered TRANSPOSED to [B,H,D,T] bf16
// so the PV GEMM's B-fragments are contiguous 32B loads.
// Scatter is branch-free: per-ni (base,step) select, inner stores are 1 madd.
// ---------------------------------------------------------------------------
__global__ void __launch_bounds__(256) qkv_gemm_kernel(
    const __bf16* __restrict__ xb, const __bf16* __restrict__ wq,
    const float* __restrict__ bias,
    __bf16* __restrict__ qd, __bf16* __restrict__ kd, __bf16* __restrict__ vd) {
  const int wave = threadIdx.x >> 5;
  const int lane = threadIdx.x & 31;
  const int l15  = lane & 15;
  const int kh   = lane >> 4;
  const int mbase = blockIdx.x * 128 + (wave & 3) * 32;
  const int nbase = blockIdx.y * 128 + (wave >> 2) * 64;
  v8f acc[2][4] = {};
  gemm_mainloop(xb, wq, mbase, nbase, DIMC, acc);

  const int bi = mbase >> 11;        // uniform: batch index of this wave tile
  const int tb = (mbase & 2047) + kh * 8;
  #pragma unroll
  for (int ni = 0; ni < 4; ++ni) {
    const int f  = nbase + ni * 16 + l15;
    const float bv = bias[f];
    const int which = f >> 10;       // uniform within a 16-wide tile
    const int fc = f & 1023;
    const int h  = fc >> 6;
    const int d  = fc & 63;
    __bf16* dst = (which == 0) ? qd : ((which == 1) ? kd : vd);
    const size_t hb = (size_t)(bi * NHEAD + h);
    const bool isv = (which == 2);
    const size_t base = isv ? (hb * HDIM + d) * SEQ         // V^T [B,H,D,T]
                            : hb * SEQ * HDIM + d;          // Q/K [B,H,T,D]
    const int step = isv ? 1 : HDIM;
    __bf16* p = dst + base;
    #pragma unroll
    for (int mi = 0; mi < 2; ++mi) {
      #pragma unroll
      for (int r = 0; r < 8; ++r) {
        const int t = tb + mi * 16 + r;
        p[(size_t)t * step] = f2bf(acc[mi][ni][r] + bv);
      }
    }
  }
}

// ---------------------------------------------------------------------------
// Flash attention with additive bias. One wave per 16 query rows,
// block = 8 waves = 128 rows, grid = (B*H, T/128). Online softmax;
// P goes through per-wave LDS for C-layout -> A-layout conversion.
// V is pre-transposed [B,H,D,T] so PV B-fragments are contiguous.
// ---------------------------------------------------------------------------
__global__ void __launch_bounds__(256) attn_kernel(
    const __bf16* __restrict__ qg, const __bf16* __restrict__ kg,
    const __bf16* __restrict__ vtg, const float* __restrict__ biasg,
    __bf16* __restrict__ og) {
  __shared__ __bf16 plds[8][16][32];
  const int wave = threadIdx.x >> 5;
  const int lane = threadIdx.x & 31;
  const int l15  = lane & 15;
  const int kh   = lane >> 4;
  const int bh = blockIdx.x;            // b*NHEAD + h
  const int b  = bh >> 4;
  const int h  = bh & 15;
  const int qbase = blockIdx.y * 128 + wave * 16;
  const __bf16* qp = qg  + (size_t)bh * SEQ * HDIM;
  const __bf16* kp = kg  + (size_t)bh * SEQ * HDIM;
  const __bf16* vt = vtg + (size_t)bh * HDIM * SEQ;   // [D][T]
  const float* brow = biasg + ((size_t)b * SEQ + qbase) * SEQ;

  // Q fragments (A-layout), d = 0..63 in two K=32 chunks
  v16bf qa[2];
  #pragma unroll
  for (int c = 0; c < 2; ++c) {
    const __bf16* ap = qp + (size_t)(qbase + l15) * HDIM + c * 32 + kh * 16;
    v8bf lo = *(const v8bf*)ap;
    v8bf hi = *(const v8bf*)(ap + 8);
    #pragma unroll
    for (int i = 0; i < 8; ++i) { qa[c][i] = lo[i]; qa[c][8 + i] = hi[i]; }
  }

  float mrun[8], lrun[8];
  #pragma unroll
  for (int r = 0; r < 8; ++r) { mrun[r] = -1e30f; lrun[r] = 0.f; }
  v8f oacc[4] = {};
  const float scale = 0.125f;            // 1/sqrt(64)

  for (int sb = 0; sb < SEQ; sb += 32) {
    // S = Q @ K^T for 32 key columns (two 16x16 tiles)
    v8f sacc[2] = {};
    #pragma unroll
    for (int ni = 0; ni < 2; ++ni) {
      const __bf16* kr = kp + (size_t)(sb + ni * 16 + l15) * HDIM + kh * 16;
      v16bf b0 = *(const v16bf*)kr;
      v16bf b1 = *(const v16bf*)(kr + 32);
      sacc[ni] = __builtin_amdgcn_wmma_f32_16x16x32_bf16(false, qa[0], false, b0, (short)0, sacc[ni], false, false);
      sacc[ni] = __builtin_amdgcn_wmma_f32_16x16x32_bf16(false, qa[1], false, b1, (short)0, sacc[ni], false, false);
    }
    // scores = S*scale + bias ; running row-max
    float sv[2][8], rmax[8];
    #pragma unroll
    for (int r = 0; r < 8; ++r) rmax[r] = -1e30f;
    #pragma unroll
    for (int ni = 0; ni < 2; ++ni) {
      const int sc = sb + ni * 16 + l15;
      #pragma unroll
      for (int r = 0; r < 8; ++r) {
        const int qr = kh * 8 + r;
        float val = sacc[ni][r] * scale + brow[(size_t)qr * SEQ + sc];
        sv[ni][r] = val;
        rmax[r] = fmaxf(rmax[r], val);
      }
    }
    #pragma unroll
    for (int msk = 1; msk < 16; msk <<= 1) {
      #pragma unroll
      for (int r = 0; r < 8; ++r)
        rmax[r] = fmaxf(rmax[r], __shfl_xor(rmax[r], msk, 32));
    }
    float corr[8], psum[8];
    #pragma unroll
    for (int r = 0; r < 8; ++r) {
      float mn = fmaxf(mrun[r], rmax[r]);
      corr[r] = __expf(mrun[r] - mn);
      mrun[r] = mn;
      psum[r] = 0.f;
    }
    // P = exp(score - m); stash to LDS (C-layout positions)
    #pragma unroll
    for (int ni = 0; ni < 2; ++ni) {
      #pragma unroll
      for (int r = 0; r < 8; ++r) {
        float p = __expf(sv[ni][r] - mrun[r]);
        psum[r] += p;
        plds[wave][kh * 8 + r][ni * 16 + l15] = f2bf(p);
      }
    }
    #pragma unroll
    for (int msk = 1; msk < 16; msk <<= 1) {
      #pragma unroll
      for (int r = 0; r < 8; ++r)
        psum[r] += __shfl_xor(psum[r], msk, 32);
    }
    #pragma unroll
    for (int r = 0; r < 8; ++r) lrun[r] = lrun[r] * corr[r] + psum[r];
    #pragma unroll
    for (int dt = 0; dt < 4; ++dt)
      #pragma unroll
      for (int r = 0; r < 8; ++r) oacc[dt][r] *= corr[r];

    // Reload P from LDS in A-layout (same-wave DS ops are in-order)
    v16bf pa;
    {
      const __bf16* lp = &plds[wave][l15][kh * 16];
      v8bf lo = *(const v8bf*)lp;
      v8bf hi = *(const v8bf*)(lp + 8);
      #pragma unroll
      for (int i = 0; i < 8; ++i) { pa[i] = lo[i]; pa[8 + i] = hi[i]; }
    }
    // O += P(16x32) @ V(32x64); V^T rows give contiguous B-fragments
    #pragma unroll
    for (int dt = 0; dt < 4; ++dt) {
      v16bf bv = *(const v16bf*)(vt + (size_t)(dt * 16 + l15) * SEQ + sb + kh * 16);
      oacc[dt] = __builtin_amdgcn_wmma_f32_16x16x32_bf16(false, pa, false, bv, (short)0, oacc[dt], false, false);
    }
  }
  // normalize and store [B,T,C] bf16
  __bf16* orow = og + ((size_t)b * SEQ + qbase) * DIMC + h * HDIM;
  #pragma unroll
  for (int dt = 0; dt < 4; ++dt) {
    #pragma unroll
    for (int r = 0; r < 8; ++r) {
      const int qr = kh * 8 + r;
      orow[(size_t)qr * DIMC + dt * 16 + l15] = f2bf(oacc[dt][r] / lrun[r]);
    }
  }
}

// ---------------------------------------------------------------------------
// Output projection: [8192,1024] x [1024,1024]^T + bias -> fp32 d_out
// ---------------------------------------------------------------------------
__global__ void __launch_bounds__(256) proj_gemm_kernel(
    const __bf16* __restrict__ ao, const __bf16* __restrict__ wo,
    const float* __restrict__ bias, float* __restrict__ out) {
  const int wave = threadIdx.x >> 5;
  const int lane = threadIdx.x & 31;
  const int l15  = lane & 15;
  const int kh   = lane >> 4;
  const int mbase = blockIdx.x * 128 + (wave & 3) * 32;
  const int nbase = blockIdx.y * 128 + (wave >> 2) * 64;
  v8f acc[2][4] = {};
  gemm_mainloop(ao, wo, mbase, nbase, DIMC, acc);
  #pragma unroll
  for (int ni = 0; ni < 4; ++ni) {
    const int n = nbase + ni * 16 + l15;
    const float bv = bias[n];
    #pragma unroll
    for (int mi = 0; mi < 2; ++mi) {
      #pragma unroll
      for (int r = 0; r < 8; ++r) {
        const int m = mbase + mi * 16 + kh * 8 + r;
        out[(size_t)m * DIMC + n] = acc[mi][ni][r] + bv;
      }
    }
  }
}

extern "C" void kernel_launch(void* const* d_in, const int* in_sizes, int n_in,
                              void* d_out, int out_size, void* d_ws, size_t ws_size,
                              hipStream_t stream) {
  (void)in_sizes; (void)n_in; (void)out_size; (void)ws_size;
  const float* x    = (const float*)d_in[0];
  const float* abias= (const float*)d_in[1];
  const float* wqkv = (const float*)d_in[2];
  const float* bqkv = (const float*)d_in[3];
  const float* wo   = (const float*)d_in[4];
  const float* bo   = (const float*)d_in[5];
  float* out = (float*)d_out;

  char* ws = (char*)d_ws;
  const size_t MB = 1024u * 1024u;
  __bf16* xb    = (__bf16*)(ws);              // 16 MB, dead after QKV gemm
  __bf16* wqkvb = (__bf16*)(ws + 16 * MB);    //  6 MB
  __bf16* wob   = (__bf16*)(ws + 22 * MB);    //  2 MB
  __bf16* qw    = (__bf16*)(ws + 24 * MB);    // 16 MB  [B,H,T,D]
  __bf16* kw    = (__bf16*)(ws + 40 * MB);    // 16 MB  [B,H,T,D]
  __bf16* vw    = (__bf16*)(ws + 56 * MB);    // 16 MB  [B,H,D,T] (transposed)
  __bf16* aow   = xb;                         // 16 MB, aliases xb (sequential)

  const int nx  = BATCH * SEQ * DIMC;   // 8388608
  const int nwq = 3 * DIMC * DIMC;      // 3145728
  const int nwo = DIMC * DIMC;          // 1048576
  cvt_f32_bf16<<<nx  / 256, 256, 0, stream>>>(x,    xb,    nx);
  cvt_f32_bf16<<<nwq / 256, 256, 0, stream>>>(wqkv, wqkvb, nwq);
  cvt_f32_bf16<<<nwo / 256, 256, 0, stream>>>(wo,   wob,   nwo);

  qkv_gemm_kernel<<<dim3(64, 24), 256, 0, stream>>>(xb, wqkvb, bqkv, qw, kw, vw);
  attn_kernel   <<<dim3(BATCH * NHEAD, SEQ / 128), 256, 0, stream>>>(qw, kw, vw, abias, aow);
  proj_gemm_kernel<<<dim3(64, 8), 256, 0, stream>>>(aow, wob, bo, out);
}
